// GalacticNavigator_10728828305815
// MI455X (gfx1250) — matrix operand order repaired and checked by reading, not compiled, not run on hardware
//
#include <hip/hip_runtime.h>
#include <hip/hip_bf16.h>

// ---------------------------------------------------------------------------
// Types for CDNA5 WMMA (wave32, 16x16x32 bf16 -> f32)
// ---------------------------------------------------------------------------
typedef __attribute__((ext_vector_type(16))) __bf16 v16bf;
typedef __attribute__((ext_vector_type(8)))  float  v8f;
typedef unsigned short u16;
typedef unsigned int   u32;

#define Bsz 8
#define Ssz 2048
#define Dh  1024
#define BS  (Bsz * Ssz)      // 16384 rows
#define NT  (Dh / 16)        // 64 column tiles
#define KS  (Dh / 32)        // 32 k-steps
#define FRAG_HALFS 512       // 32 lanes * 16 halfs = 1KB per packed B fragment
#define RWG 64               // recurrence workgroups (one column tile each)

static __device__ __forceinline__ u16 f2bf(float f) {
    u32 u = __float_as_uint(f);
    u32 r = (u + 0x7FFFu + ((u >> 16) & 1u)) >> 16;   // round-to-nearest-even
    return (u16)r;
}
static __device__ __forceinline__ float bf2f(u16 s) {
    return __uint_as_float(((u32)s) << 16);
}

// A-matrix fragment, 16x32 bf16, row-major source (lda in elements).
// ISA layout: row = lane&15; VGPR v holds K = (v>=4?16:0) + (lane>>4)*8 + (v&3)*2 + h
// -> two contiguous 16B runs per lane => 2x b128.
static __device__ __forceinline__ v16bf load_a_frag(const u16* A, int lda, int lane) {
    union { v16bf v; u32 u[8]; } r;
    int row = lane & 15, hi = lane >> 4;
#pragma unroll
    for (int v = 0; v < 8; ++v) {
        int k = ((v & 4) << 2) + (hi << 3) + ((v & 3) << 1);
        r.u[v] = *(const u32*)(A + row * lda + k);
    }
    return r.v;
}

// B-matrix fragment from PRE-PACKED layout (global or LDS): 1KB block,
// lane's 16 halfs contiguous at P + lane*16  =>  2x b128 (global or ds).
static __device__ __forceinline__ v16bf load_b_packed(const u16* P, int lane) {
    union { v16bf v; u32 u[8]; } r;
    const u32* p = (const u32*)(P + lane * 16);
#pragma unroll
    for (int i = 0; i < 8; ++i) r.u[i] = p[i];
    return r.v;
}

// element offset of packed fragment (gate g, column tile nt, k-step ks)
static __device__ __forceinline__ size_t frag_off(int g, int nt, int ks) {
    return ((size_t)((g * NT + nt) * KS + ks)) * FRAG_HALFS;
}

static __device__ __forceinline__ v8f wmma_bf16(v16bf a, v16bf b, v8f c) {
    return __builtin_amdgcn_wmma_f32_16x16x32_bf16(false, a, false, b, (short)0, c, false, false);
}

static __device__ __forceinline__ float sigm(float x) { return 1.0f / (1.0f + __expf(-x)); }

// ---------------------------------------------------------------------------
// Kernel 1: split + cast GRU weights and pack into WMMA B-fragment layout.
// ---------------------------------------------------------------------------
__global__ void gn_pack_weights(const float* __restrict__ Wr, const float* __restrict__ Wu,
                                const float* __restrict__ Wc, u16* __restrict__ WxP,
                                u16* __restrict__ WhP) {
    int idx = blockIdx.x * blockDim.x + threadIdx.x;      // over Dh*Dh (k,n)
    if (idx >= Dh * Dh) return;
    int k = idx >> 10, n = idx & (Dh - 1);
    int nt = n >> 4, col = n & 15;
    int ks = k >> 5, kin = k & 31;
    int lane = (((kin >> 3) & 1) << 4) | col;
    int e = (kin & 7) | ((kin >> 4) << 3);
    const float* Ws[3] = {Wr, Wu, Wc};
#pragma unroll
    for (int g = 0; g < 3; ++g) {
        size_t p = frag_off(g, nt, ks) + lane * 16 + e;
        WxP[p] = f2bf(Ws[g][(size_t)k * Dh + n]);
        WhP[p] = f2bf(Ws[g][(size_t)(Dh + k) * Dh + n]);
    }
}

// ---------------------------------------------------------------------------
// Kernel 2: embedding gather + bf16 cast -> X [BS, 1024]
// ---------------------------------------------------------------------------
__global__ void gn_embed(const int* __restrict__ ids, const float* __restrict__ er,
                         const float* __restrict__ ei, u16* __restrict__ X) {
    int idx = blockIdx.x * blockDim.x + threadIdx.x;      // over BS*512
    if (idx >= BS * 512) return;
    int j = idx & 511, bs = idx >> 9;
    int tok = ids[bs];
    X[(size_t)bs * Dh + j]       = f2bf(er[(size_t)tok * 512 + j]);
    X[(size_t)bs * Dh + 512 + j] = f2bf(ei[(size_t)tok * 512 + j]);
}

// ---------------------------------------------------------------------------
// Kernel 3: Gx[g] = X @ Wx[g] + b[g]. 2x2 WMMA tiles per wave (unchanged;
// round-2 asm confirmed clean b128 + 4x v_wmma pipelined inner loop).
// ---------------------------------------------------------------------------
__global__ __launch_bounds__(256) void gn_gemm_gx(const u16* __restrict__ X,
                                                  const u16* __restrict__ WxP,
                                                  const float* __restrict__ br,
                                                  const float* __restrict__ bu,
                                                  const float* __restrict__ bc,
                                                  float* __restrict__ Gx) {
    int wave = (blockIdx.x * blockDim.x + threadIdx.x) >> 5;
    int lane = threadIdx.x & 31;
    int nb = wave & 31;            // N block of 32 cols
    int mb = (wave >> 5) & 511;    // M block of 32 rows
    int g  = wave >> 14;           // 0..2

    const u16* A0 = X + (size_t)(mb * 2 + 0) * 16 * Dh;
    const u16* A1 = X + (size_t)(mb * 2 + 1) * 16 * Dh;
    const int n0 = nb * 2, n1 = nb * 2 + 1;

    v8f acc00 = {}, acc01 = {}, acc10 = {}, acc11 = {};
#pragma unroll 2
    for (int ks = 0; ks < KS; ++ks) {
        const int k0 = ks * 32;
        v16bf a0 = load_a_frag(A0 + k0, Dh, lane);
        v16bf a1 = load_a_frag(A1 + k0, Dh, lane);
        v16bf b0 = load_b_packed(WxP + frag_off(g, n0, ks), lane);
        v16bf b1 = load_b_packed(WxP + frag_off(g, n1, ks), lane);
        acc00 = wmma_bf16(a0, b0, acc00);
        acc01 = wmma_bf16(a0, b1, acc01);
        acc10 = wmma_bf16(a1, b0, acc10);
        acc11 = wmma_bf16(a1, b1, acc11);
    }
    const float* bias = (g == 0) ? br : (g == 1) ? bu : bc;
    float* dst = Gx + (size_t)g * BS * Dh;
    const int colLo = lane & 15;
    const int rowHi = (lane >> 4) << 3;
#pragma unroll
    for (int v = 0; v < 8; ++v) {
        int m0 = (mb * 2 + 0) * 16 + rowHi + v;
        int m1 = (mb * 2 + 1) * 16 + rowHi + v;
        int c0 = n0 * 16 + colLo, c1 = n1 * 16 + colLo;
        dst[(size_t)m0 * Dh + c0] = acc00[v] + bias[c0];
        dst[(size_t)m0 * Dh + c1] = acc01[v] + bias[c1];
        dst[(size_t)m1 * Dh + c0] = acc10[v] + bias[c0];
        dst[(size_t)m1 * Dh + c1] = acc11[v] + bias[c1];
    }
}

// ---------------------------------------------------------------------------
// Kernel 4: zero shared recurrence state (h, r*h, barrier counter) per launch.
// ---------------------------------------------------------------------------
__global__ void gn_init_state(u32* __restrict__ hG, u32* __restrict__ rhG,
                              u32* __restrict__ cnt) {
    int idx = blockIdx.x * blockDim.x + threadIdx.x;   // over 8192 u32 each
    if (idx < 16 * Dh / 2) { hG[idx] = 0; rhG[idx] = 0; }
    if (idx == 0) *cnt = 0;
}

// ---------------------------------------------------------------------------
// Kernel 5: GRU recurrence, 64 WGs (one WGP each), one 16-col tile per WG.
// The WG's W_h slice (3 gates x 32 ks x 1KB = 96KB) is preloaded into LDS,
// so the 2048 serial steps generate ZERO weight traffic (the old design
// streamed 6MB/step through one WGP's L2 port -> hopeless). Shared state
// per step is just h / r*h (32KB bf16 each, L2-broadcast) + 2 device
// barriers. K split across waves (4-way r,u / 8-way c) keeps each wave's
// WMMA accumulate chain to 8 / 4 -> short per-step critical path.
// Dynamic LDS: 96KB weights | 8KB partials | 1KB u-gate.
// ---------------------------------------------------------------------------
__global__ __launch_bounds__(256, 1) void gn_recurrence(const float* __restrict__ Gx,
                                                        const u16* __restrict__ WhP,
                                                        u16* __restrict__ hG,
                                                        u16* __restrict__ rhG,
                                                        u32* __restrict__ cnt,
                                                        float* __restrict__ hs) {
    extern __shared__ __align__(16) char smem[];
    u16*   Wl = (u16*)smem;                        // [3][KS][FRAG_HALFS]
    float* pl = (float*)(smem + 3 * KS * FRAG_HALFS * 2);   // [8][256] partials
    float* uv = (float*)(smem + 3 * KS * FRAG_HALFS * 2 + 8 * 256 * 4); // [256]

    const int tid = threadIdx.x, lane = tid & 31, wave = tid >> 5;
    const int nt = blockIdx.x;                     // owned column tile
    const float* Gr = Gx;
    const float* Gu = Gx + (size_t)BS * Dh;
    const float* Gc = Gx + (size_t)2 * BS * Dh;

    // ---- one-time: W_h slice -> LDS (3 x 32KB contiguous packed regions)
#pragma unroll
    for (int g = 0; g < 3; ++g) {
        const uint4* src = (const uint4*)(WhP + frag_off(g, nt, 0));
        uint4* dst = (uint4*)(Wl + g * KS * FRAG_HALFS);
        for (int i = tid; i < KS * FRAG_HALFS / 8; i += 256) dst[i] = src[i];
    }
    __syncthreads();

    const int eM = tid >> 4, eC = tid & 15;        // combine-phase mapping
    const int eCol = nt * 16 + eC;
    u32 barTarget = RWG;

    for (int t = 0; t < Ssz; ++t) {
        // ---- phase 1: r (gate 0) and u (gate 1) ------------------------
        {
            const int gi = wave & 1;               // gate
            const int q  = wave >> 1;              // k-slice (8 ks each)
            v8f acc = {};
            for (int ks = q * 8; ks < q * 8 + 8; ++ks) {
                v16bf a = load_a_frag(hG + ks * 32, Dh, lane);
                v16bf b = load_b_packed(Wl + (gi * KS + ks) * FRAG_HALFS, lane);
                acc = wmma_bf16(a, b, acc);
            }
            float* p = pl + (gi * 4 + q) * 256;
            const int colLo = lane & 15, rowHi = (lane >> 4) << 3;
#pragma unroll
            for (int v = 0; v < 8; ++v) p[(rowHi + v) * 16 + colLo] = acc[v];
        }
        __syncthreads();
        if (eM < Bsz) {                            // combine + activations
            float s0 = 0.f, s1 = 0.f;
#pragma unroll
            for (int q = 0; q < 4; ++q) {
                s0 += pl[(0 * 4 + q) * 256 + tid];
                s1 += pl[(1 * 4 + q) * 256 + tid];
            }
            size_t gx = (size_t)(eM * Ssz + t) * Dh + eCol;
            float r = sigm(s0 + Gr[gx]);
            uv[tid] = sigm(s1 + Gu[gx]);
            rhG[eM * Dh + eCol] = f2bf(r * bf2f(hG[eM * Dh + eCol]));
        }
        // ---- device barrier #1 (rh visible everywhere) -----------------
        __syncthreads();
        if (tid == 0) {
            __hip_atomic_fetch_add(cnt, 1u, __ATOMIC_ACQ_REL, __HIP_MEMORY_SCOPE_AGENT);
            while (__hip_atomic_load(cnt, __ATOMIC_ACQUIRE, __HIP_MEMORY_SCOPE_AGENT) < barTarget)
                __builtin_amdgcn_s_sleep(1);
        }
        barTarget += RWG;
        __syncthreads();

        // ---- phase 2: candidate c --------------------------------------
        {
            const int q = wave;                    // k-slice (4 ks each)
            v8f acc = {};
            for (int ks = q * 4; ks < q * 4 + 4; ++ks) {
                v16bf a = load_a_frag(rhG + ks * 32, Dh, lane);
                v16bf b = load_b_packed(Wl + (2 * KS + ks) * FRAG_HALFS, lane);
                acc = wmma_bf16(a, b, acc);
            }
            float* p = pl + q * 256;
            const int colLo = lane & 15, rowHi = (lane >> 4) << 3;
#pragma unroll
            for (int v = 0; v < 8; ++v) p[(rowHi + v) * 16 + colLo] = acc[v];
        }
        __syncthreads();
        if (eM < Bsz) {                            // combine + state update
            float s = 0.f;
#pragma unroll
            for (int q = 0; q < 8; ++q) s += pl[q * 256 + tid];
            float c = tanhf(s + Gc[(size_t)(eM * Ssz + t) * Dh + eCol]);
            float u = uv[tid];
            float hn = (1.0f - u) * bf2f(hG[eM * Dh + eCol]) + u * c;
            hG[eM * Dh + eCol] = f2bf(hn);
            hs[(size_t)(eM * Ssz + t) * Dh + eCol] = hn;
        }
        // ---- device barrier #2 (h visible everywhere) ------------------
        __syncthreads();
        if (tid == 0) {
            __hip_atomic_fetch_add(cnt, 1u, __ATOMIC_ACQ_REL, __HIP_MEMORY_SCOPE_AGENT);
            while (__hip_atomic_load(cnt, __ATOMIC_ACQUIRE, __HIP_MEMORY_SCOPE_AGENT) < barTarget)
                __builtin_amdgcn_s_sleep(1);
        }
        barTarget += RWG;
        __syncthreads();
    }
}

// ---------------------------------------------------------------------------
// Kernel 6: 6 LayerNorms over the two 512-wide halves. One row per block.
// ---------------------------------------------------------------------------
__global__ __launch_bounds__(256) void gn_layernorm(const float* __restrict__ hs,
                                                    const float* __restrict__ scale,
                                                    const float* __restrict__ bias,
                                                    float* __restrict__ out) {
    const int row = blockIdx.x;          // b*S + s, 0..16383
    const int tid = threadIdx.x;
    __shared__ float sm[Dh];
    __shared__ float rs[256], rs2[256];

    for (int i = tid; i < Dh; i += 256) sm[i] = hs[(size_t)row * Dh + i];
    __syncthreads();

    for (int hf = 0; hf < 2; ++hf) {
        float s = 0.f, s2 = 0.f;
        for (int i = tid; i < 512; i += 256) {
            float x = sm[hf * 512 + i];
            s += x; s2 += x * x;
        }
        rs[tid] = s; rs2[tid] = s2;
        __syncthreads();
        for (int o = 128; o > 0; o >>= 1) {
            if (tid < o) { rs[tid] += rs[tid + o]; rs2[tid] += rs2[tid + o]; }
            __syncthreads();
        }
        float mu  = rs[0] * (1.0f / 512.0f);
        float var = rs2[0] * (1.0f / 512.0f) - mu * mu;
        float inv = rsqrtf(var + 1e-6f);
        __syncthreads();
        for (int o = hf; o < 6; o += 2) {
            for (int i = tid; i < 512; i += 256) {
                float x = (sm[hf * 512 + i] - mu) * inv;
                out[(size_t)o * BS * 512 + (size_t)row * 512 + i] =
                    x * scale[o * 512 + i] + bias[o * 512 + i];
            }
        }
        __syncthreads();
    }
}

// ---------------------------------------------------------------------------
// Host launcher
// ---------------------------------------------------------------------------
extern "C" void kernel_launch(void* const* d_in, const int* in_sizes, int n_in,
                              void* d_out, int out_size, void* d_ws, size_t ws_size,
                              hipStream_t stream) {
    (void)in_sizes; (void)n_in; (void)out_size; (void)ws_size;
    const int*   token_ids = (const int*)  d_in[0];
    const float* emb_r     = (const float*)d_in[1];
    const float* emb_i     = (const float*)d_in[2];
    const float* W_r       = (const float*)d_in[3];
    const float* b_r       = (const float*)d_in[4];
    const float* W_u       = (const float*)d_in[5];
    const float* b_u       = (const float*)d_in[6];
    const float* W_c       = (const float*)d_in[7];
    const float* b_c       = (const float*)d_in[8];
    const float* ln_scale  = (const float*)d_in[9];
    const float* ln_bias   = (const float*)d_in[10];
    float* out = (float*)d_out;

    // Workspace layout
    char* ws = (char*)d_ws;
    size_t off = 0;
    u16*   X   = (u16*)(ws + off);  off += (size_t)BS * Dh * 2;          // 32 MB
    u16*   WxP = (u16*)(ws + off);  off += (size_t)3 * Dh * Dh * 2;      // 6 MB
    u16*   WhP = (u16*)(ws + off);  off += (size_t)3 * Dh * Dh * 2;      // 6 MB
    float* Gx  = (float*)(ws + off); off += (size_t)3 * BS * Dh * 4;     // 192 MB
    float* hs  = (float*)(ws + off); off += (size_t)BS * Dh * 4;         // 64 MB
    u16*   hG  = (u16*)(ws + off);  off += (size_t)16 * Dh * 2;          // 32 KB
    u16*   rhG = (u16*)(ws + off);  off += (size_t)16 * Dh * 2;          // 32 KB
    u32*   cnt = (u32*)(ws + off);  off += 256;

    // 1) weight split + bf16 cast + B-fragment packing
    gn_pack_weights<<<(Dh * Dh) / 256, 256, 0, stream>>>(W_r, W_u, W_c, WxP, WhP);
    // 2) embedding gather
    gn_embed<<<(BS * 512) / 256, 256, 0, stream>>>(token_ids, emb_r, emb_i, X);
    // 3) Gx GEMM: 49152 waves (2x2 tiles each), 8 waves/block
    gn_gemm_gx<<<6144, 256, 0, stream>>>(X, WxP, b_r, b_u, b_c, Gx);
    // 4) zero h / r*h / barrier counter (every launch -> deterministic replay)
    gn_init_state<<<32, 256, 0, stream>>>((u32*)hG, (u32*)rhG, cnt);
    // 5) recurrence: 64 WGs, LDS-resident weights (96KB) + 9KB scratch
    size_t lds = (size_t)3 * KS * FRAG_HALFS * 2 + 8 * 256 * 4 + 256 * 4;
    gn_recurrence<<<RWG, 256, lds, stream>>>(Gx, WhP, hG, rhG, cnt, hs);
    // 6) layernorms -> output [6, B, S, 512]
    gn_layernorm<<<BS, 256, 0, stream>>>(hs, ln_scale, ln_bias, out);
}
